// Decoder_83803401879968
// MI455X (gfx1250) — compile-verified, matrix operand order checked by reading, not run on hardware
//
#include <hip/hip_runtime.h>
#include <cstdint>
#include <cstddef>

typedef __bf16 bf16_t;
typedef bf16_t v16bf __attribute__((ext_vector_type(16)));
typedef bf16_t v8bf  __attribute__((ext_vector_type(8)));
typedef float  v8f   __attribute__((ext_vector_type(8)));

#define BB   128
#define PP   196
#define ENCD 2048
#define DECD 512
#define EMBD 512
#define VV   10000
#define TCAP 30
#define TM   29

__device__ __forceinline__ float sigf(float x) { return 1.0f / (1.0f + __expf(-x)); }

// Load one 16-element bf16 WMMA fragment half-pair: {p[0..7], p[16..23]}
// (caller pre-offsets p by hsel*8 so lanes 16..31 pick the {8..15, 24..31} K sub-block,
//  matching the ISA 7.12.2 16-bit A/B 16x32 VGPR layout).
__device__ __forceinline__ v16bf load_frag(const bf16_t* __restrict__ p)
{
    v8bf lo = *(const v8bf*)(p);
    v8bf hi = *(const v8bf*)(p + 16);
    v16bf v;
#pragma unroll
    for (int i = 0; i < 8; ++i) { v[i] = lo[i]; v[8 + i] = hi[i]; }
    return v;
}

__device__ __forceinline__ v8f wmma_bf16(v16bf a, v16bf b, v8f c)
{
    return __builtin_amdgcn_wmma_f32_16x16x32_bf16(false, a, false, b, (short)0, c, false, false);
}

// -------------------------------------------------------------------------
// WMMA GEMM: C[M,N] = act( A[M,K] @ W[N,K]^T + bias[N] )   (bf16 in, f32 out)
// 8 waves/block, 4(M)x2(N) wave grid; each wave owns a 16 x (16*NT) C tile
// (NT accumulators share one A fragment). Software-pipelined K loop.
// NT==4 requires N % 128 == 0 (host guarantees); NT==2 handles the ragged edge
// with a wave-uniform (scalar) tile count.
// ACT: 0=none 1=sigmoid. BETA: accumulate into C. MASKED: C *= (t < lenDec[row]).
// -------------------------------------------------------------------------
template<int NT, int ACT, int BETA, int MASKED>
__global__ void wmma_gemm(const bf16_t* __restrict__ A, const bf16_t* __restrict__ W,
                          const float* __restrict__ bias, float* __restrict__ C,
                          int M, int N, int K, int ldc,
                          const int* __restrict__ lenDec, int tstep)
{
    // force wave-uniform values into SGPRs -> scalar branches, no EXEC juggling
    const int wave = __builtin_amdgcn_readfirstlane((int)(threadIdx.x >> 5));
    const int lane = threadIdx.x & 31;
    const int hsel = lane >> 4;
    const int r    = lane & 15;

    const int m0 = (blockIdx.y * 4 + (wave >> 1)) * 16;
    const int n0 = (blockIdx.x * 2 + (wave & 1)) * (16 * NT);
    if (m0 >= M || n0 >= N) return;

    const bf16_t* aRow = A + (size_t)(m0 + r) * K + hsel * 8;

    v8f acc[NT];
    const v8f vzero = {};
#pragma unroll
    for (int t = 0; t < NT; ++t) acc[t] = vzero;

    int ntiles = NT;
    if (NT == 2 && (n0 + 16) >= N) ntiles = 1;

    if (NT == 2 && ntiles == 1) {
        // ragged edge: single 16x16 tile, pipelined
        const bf16_t* w0 = W + (size_t)(n0 + r) * K + hsel * 8;
        v16bf av = load_frag(aRow);
        v16bf bv = load_frag(w0);
        for (int k = 32; k < K; k += 32) {
            v16bf an = load_frag(aRow + k);
            v16bf bn = load_frag(w0 + k);
            acc[0] = wmma_bf16(av, bv, acc[0]);
            av = an; bv = bn;
        }
        acc[0] = wmma_bf16(av, bv, acc[0]);
    } else {
        const bf16_t* w[NT];
#pragma unroll
        for (int t = 0; t < NT; ++t)
            w[t] = W + (size_t)(n0 + 16 * t + r) * K + hsel * 8;

        v16bf av = load_frag(aRow);
        v16bf bv[NT];
#pragma unroll
        for (int t = 0; t < NT; ++t) bv[t] = load_frag(w[t]);

        for (int k = 32; k < K; k += 32) {
            // issue next iteration's loads before consuming current fragments
            v16bf an = load_frag(aRow + k);
            v16bf bn[NT];
#pragma unroll
            for (int t = 0; t < NT; ++t) bn[t] = load_frag(w[t] + k);
#pragma unroll
            for (int t = 0; t < NT; ++t) acc[t] = wmma_bf16(av, bv[t], acc[t]);
            av = an;
#pragma unroll
            for (int t = 0; t < NT; ++t) bv[t] = bn[t];
        }
#pragma unroll
        for (int t = 0; t < NT; ++t) acc[t] = wmma_bf16(av, bv[t], acc[t]);
    }

    // C/D layout: VGPR j -> M = m0 + hsel*8 + j, N = n0 + 16*t + r
#pragma unroll
    for (int t = 0; t < NT; ++t) {
        if (NT == 2 && t == 1 && ntiles == 1) break;
        const int col = n0 + t * 16 + r;
        const float bval = bias ? bias[col] : 0.0f;
#pragma unroll
        for (int j = 0; j < 8; ++j) {
            const int row = m0 + hsel * 8 + j;
            float v = acc[t][j] + bval;
            if (ACT == 1) v = sigf(v);
            float* cp = C + (size_t)row * ldc + col;
            if (BETA) v += *cp;
            if (MASKED) v *= (tstep < lenDec[row]) ? 1.0f : 0.0f;
            *cp = v;
        }
    }
}

// ------------------- support kernels -------------------

// stable argsort of -lens over B=128 (rank counting), emit sorted caps/len_dec/order
__global__ void k_order(const int* __restrict__ lenCaps, const int* __restrict__ encCaps,
                        int* __restrict__ ordW, int* __restrict__ lendecW, int* __restrict__ capsS,
                        float* __restrict__ outCaps, float* __restrict__ outLenDec,
                        float* __restrict__ outOrder)
{
    __shared__ int lens[BB];
    __shared__ int ordSh[BB];
    const int i = threadIdx.x;
    lens[i] = lenCaps[i];
    __syncthreads();
    const int li = lens[i];
    int rank = 0;
    for (int j = 0; j < BB; ++j) {
        const int lj = lens[j];
        if (lj > li || (lj == li && j < i)) rank++;
    }
    ordSh[rank] = i;
    __syncthreads();
    const int src = ordSh[i];
    ordW[i] = src;
    const int ld = lens[src] - 1;
    lendecW[i]  = ld;
    outLenDec[i] = (float)ld;
    outOrder[i]  = (float)src;
    for (int t = 0; t < TCAP; ++t) {
        const int cv = encCaps[src * TCAP + t];
        capsS[i * TCAP + t] = cv;
        outCaps[i * TCAP + t] = (float)cv;
    }
}

// permute enc_out by order and convert to bf16
__global__ void k_permcvt(const float* __restrict__ encOut, const int* __restrict__ ordW,
                          bf16_t* __restrict__ encBf, size_t total)
{
    size_t idx = (size_t)blockIdx.x * blockDim.x + threadIdx.x;
    if (idx >= total) return;
    const size_t per = (size_t)PP * ENCD;
    const int b = (int)(idx / per);
    const size_t rem = idx - (size_t)b * per;
    encBf[idx] = (bf16_t)encOut[(size_t)ordW[b] * per + rem];
}

// mu[b,d] = mean_p enc[b,p,d]  (bf16 out for the init GEMMs)
__global__ void k_mu(const bf16_t* __restrict__ encBf, bf16_t* __restrict__ muBf)
{
    const int d = blockIdx.x * 256 + threadIdx.x;
    const int b = blockIdx.y;
    const bf16_t* eb = encBf + (size_t)b * PP * ENCD + d;
    float s = 0.0f;
    for (int p = 0; p < PP; ++p) s += (float)eb[(size_t)p * ENCD];
    muBf[b * ENCD + d] = (bf16_t)(s * (1.0f / (float)PP));
}

// gather embeddings for sorted caps -> bf16 [B][TM][EMBD]
__global__ void k_embed(const float* __restrict__ emb, const int* __restrict__ capsS,
                        bf16_t* __restrict__ embeds)
{
    const int idx = blockIdx.x * 256 + threadIdx.x;
    if (idx >= BB * TM * EMBD) return;
    const int d = idx & (EMBD - 1);
    const int bt = idx >> 9;
    const int t = bt % TM;
    const int b = bt / TM;
    const int tok = capsS[b * TCAP + t];
    embeds[idx] = (bf16_t)emb[(size_t)tok * EMBD + d];
}

// generic f32 -> bf16 convert
__global__ void k_cvt(const float* __restrict__ src, bf16_t* __restrict__ dst, size_t n)
{
    size_t i = (size_t)blockIdx.x * blockDim.x + threadIdx.x;
    if (i < n) dst[i] = (bf16_t)src[i];
}

// e[b,p] = dot(relu(att_enc[b,p,:] + att_dec[b,:]), W_att[0,:]) + b_att  (one wave per (b,p))
__global__ void k_e(const float* __restrict__ attEnc, const float* __restrict__ attDec,
                    const float* __restrict__ Watt, const float* __restrict__ bAtt,
                    float* __restrict__ e)
{
    const int widx = (int)((blockIdx.x * (size_t)blockDim.x + threadIdx.x) >> 5);
    const int lane = threadIdx.x & 31;
    if (widx >= BB * PP) return;
    const int b = widx / PP;
    const float* ae = attEnc + (size_t)widx * DECD;
    const float* ad = attDec + (size_t)b * DECD;
    float s = 0.0f;
    for (int d = lane; d < DECD; d += 32) {
        float v = ae[d] + ad[d];
        v = v > 0.0f ? v : 0.0f;
        s += v * Watt[d];
    }
    for (int off = 16; off > 0; off >>= 1) s += __shfl_xor(s, off, 32);
    if (lane == 0) e[widx] = s + bAtt[0];
}

// softmax over P=196 per batch; also emit masked alphas to d_out
__global__ void k_softmax(const float* __restrict__ e, float* __restrict__ alpha,
                          float* __restrict__ outAlphas, const int* __restrict__ lenDec, int t)
{
    __shared__ float sm[256];
    const int b = blockIdx.x;
    const int tid = threadIdx.x;
    const float* eb = e + b * PP;
    const float v = (tid < PP) ? eb[tid] : -3.0e38f;
    sm[tid] = v;
    __syncthreads();
    for (int s = 128; s > 0; s >>= 1) {
        if (tid < s) sm[tid] = fmaxf(sm[tid], sm[tid + s]);
        __syncthreads();
    }
    const float mx = sm[0];
    __syncthreads();
    const float ex = (tid < PP) ? __expf(v - mx) : 0.0f;
    sm[tid] = ex;
    __syncthreads();
    for (int s = 128; s > 0; s >>= 1) {
        if (tid < s) sm[tid] += sm[tid + s];
        __syncthreads();
    }
    const float inv = 1.0f / sm[0];
    if (tid < PP) {
        const float a = ex * inv;
        alpha[b * PP + tid] = a;
        const float m = (t < lenDec[b]) ? 1.0f : 0.0f;
        outAlphas[((size_t)b * TM + t) * PP + tid] = a * m;
    }
}

// weighted[b,d] = sum_p enc[b,p,d] * alpha[b,p]
__global__ void k_weighted(const bf16_t* __restrict__ encBf, const float* __restrict__ alpha,
                           float* __restrict__ weighted)
{
    const int d = blockIdx.x * 256 + threadIdx.x;
    const int b = blockIdx.y;
    const bf16_t* eb = encBf + (size_t)b * PP * ENCD + d;
    const float* ab = alpha + b * PP;
    float s = 0.0f;
    for (int p = 0; p < PP; ++p) s += (float)eb[(size_t)p * ENCD] * ab[p];
    weighted[b * ENCD + d] = s;
}

// x = concat(emb_t, weighted*gate) in bf16
__global__ void k_wgx(const bf16_t* __restrict__ embeds, const float* __restrict__ weighted,
                      const float* __restrict__ gate, bf16_t* __restrict__ xbf, int t)
{
    const int d = blockIdx.x * 256 + threadIdx.x;   // 0..2559
    const int b = blockIdx.y;
    if (d < EMBD) {
        xbf[(size_t)b * (EMBD + ENCD) + d] = embeds[((size_t)b * TM + t) * EMBD + d];
    } else {
        const int dd = d - EMBD;
        xbf[(size_t)b * (EMBD + ENCD) + d] =
            (bf16_t)(weighted[b * ENCD + dd] * gate[b * ENCD + dd]);
    }
}

// LSTM pointwise + masked state update; emits bf16 h (masked) and h_new
__global__ void k_lstm(const float* __restrict__ gates, float* __restrict__ h, float* __restrict__ c,
                       bf16_t* __restrict__ hbf, bf16_t* __restrict__ hnewbf,
                       const int* __restrict__ lenDec, int t)
{
    const int j = blockIdx.x * 256 + threadIdx.x;   // 0..511
    const int b = blockIdx.y;
    if (j >= DECD) return;
    const float* g = gates + (size_t)b * (4 * DECD);
    const float iv = sigf(g[j]);
    const float fv = sigf(g[DECD + j]);
    const float gv = tanhf(g[2 * DECD + j]);
    const float ov = sigf(g[3 * DECD + j]);
    const int o = b * DECD + j;
    const float cn = fv * c[o] + iv * gv;
    const float hn = ov * tanhf(cn);
    hnewbf[o] = (bf16_t)hn;
    const bool m = t < lenDec[b];
    const float hq = m ? hn : h[o];
    const float cq = m ? cn : c[o];
    h[o] = hq;
    c[o] = cq;
    hbf[o] = (bf16_t)hq;
}

// ------------------- host side -------------------

static inline void launch_gemm(const bf16_t* A, const bf16_t* W, const float* bias, float* C,
                               int M, int N, int K, int ldc, int act, int beta,
                               const int* lenDec, int t, hipStream_t s)
{
    const bool wide = (N % 128) == 0;   // NT=4 path needs N % 128 == 0
    const int ntw = wide ? 4 : 2;
    dim3 grd((unsigned)((N + 32 * ntw - 1) / (32 * ntw)), (unsigned)((M + 63) / 64));
    dim3 blk(256);
    if (act == 1)
        hipLaunchKernelGGL((wmma_gemm<4, 1, 0, 0>), grd, blk, 0, s, A, W, bias, C, M, N, K, ldc, (const int*)nullptr, 0);
    else if (beta)
        hipLaunchKernelGGL((wmma_gemm<4, 0, 1, 0>), grd, blk, 0, s, A, W, bias, C, M, N, K, ldc, (const int*)nullptr, 0);
    else if (lenDec)
        hipLaunchKernelGGL((wmma_gemm<2, 0, 0, 1>), grd, blk, 0, s, A, W, bias, C, M, N, K, ldc, lenDec, t);
    else if (wide)
        hipLaunchKernelGGL((wmma_gemm<4, 0, 0, 0>), grd, blk, 0, s, A, W, bias, C, M, N, K, ldc, (const int*)nullptr, 0);
    else
        hipLaunchKernelGGL((wmma_gemm<2, 0, 0, 0>), grd, blk, 0, s, A, W, bias, C, M, N, K, ldc, (const int*)nullptr, 0);
}

extern "C" void kernel_launch(void* const* d_in, const int* in_sizes, int n_in,
                              void* d_out, int out_size, void* d_ws, size_t ws_size,
                              hipStream_t stream)
{
    const float* enc_out   = (const float*)d_in[0];
    const int*   enc_caps  = (const int*)d_in[1];
    const int*   len_caps  = (const int*)d_in[2];
    const float* emb       = (const float*)d_in[3];
    const float* W_enc_att = (const float*)d_in[4];
    const float* b_enc_att = (const float*)d_in[5];
    const float* W_dec_att = (const float*)d_in[6];
    const float* b_dec_att = (const float*)d_in[7];
    const float* W_att     = (const float*)d_in[8];
    const float* b_att     = (const float*)d_in[9];
    const float* W_init_h  = (const float*)d_in[10];
    const float* b_init_h  = (const float*)d_in[11];
    const float* W_init_c  = (const float*)d_in[12];
    const float* b_init_c  = (const float*)d_in[13];
    const float* W_ih      = (const float*)d_in[14];
    const float* b_ih      = (const float*)d_in[15];
    const float* W_hh      = (const float*)d_in[16];
    const float* b_hh      = (const float*)d_in[17];
    const float* W_fbeta   = (const float*)d_in[18];
    const float* b_fbeta   = (const float*)d_in[19];
    const float* W_fc      = (const float*)d_in[20];
    const float* b_fc      = (const float*)d_in[21];

    char* ws = (char*)d_ws;
    size_t off = 0;
    auto take = [&](size_t bytes) -> char* {
        char* p = ws + off;
        off = (off + bytes + 255) & ~(size_t)255;
        return p;
    };

    int*    ordW    = (int*)take(BB * 4);
    int*    lendecW = (int*)take(BB * 4);
    int*    capsS   = (int*)take(BB * TCAP * 4);
    bf16_t* encBf   = (bf16_t*)take((size_t)BB * PP * ENCD * 2);
    float*  attEnc  = (float*)take((size_t)BB * PP * DECD * 4);
    bf16_t* muBf    = (bf16_t*)take((size_t)BB * ENCD * 2);
    bf16_t* embeds  = (bf16_t*)take((size_t)BB * TM * EMBD * 2);
    bf16_t* wEncAtt = (bf16_t*)take((size_t)DECD * ENCD * 2);
    bf16_t* wDecAtt = (bf16_t*)take((size_t)DECD * DECD * 2);
    bf16_t* wInitH  = (bf16_t*)take((size_t)DECD * ENCD * 2);
    bf16_t* wInitC  = (bf16_t*)take((size_t)DECD * ENCD * 2);
    bf16_t* wIh     = (bf16_t*)take((size_t)(4 * DECD) * (EMBD + ENCD) * 2);
    bf16_t* wHh     = (bf16_t*)take((size_t)(4 * DECD) * DECD * 2);
    bf16_t* wFbeta  = (bf16_t*)take((size_t)ENCD * DECD * 2);
    bf16_t* wFc     = (bf16_t*)take((size_t)VV * DECD * 2);
    float*  hS      = (float*)take((size_t)BB * DECD * 4);
    float*  cS      = (float*)take((size_t)BB * DECD * 4);
    bf16_t* hbf     = (bf16_t*)take((size_t)BB * DECD * 2);
    bf16_t* hnewbf  = (bf16_t*)take((size_t)BB * DECD * 2);
    float*  attDec  = (float*)take((size_t)BB * DECD * 4);
    float*  eBuf    = (float*)take((size_t)BB * PP * 4);
    float*  alpha   = (float*)take((size_t)BB * PP * 4);
    float*  weighted= (float*)take((size_t)BB * ENCD * 4);
    float*  gate    = (float*)take((size_t)BB * ENCD * 4);
    bf16_t* xbf     = (bf16_t*)take((size_t)BB * (EMBD + ENCD) * 2);
    float*  gatesB  = (float*)take((size_t)BB * 4 * DECD * 4);

    float* out       = (float*)d_out;
    float* outCaps   = out;                                   // 128*30
    float* outLenDec = out + (size_t)BB * TCAP;               // 128
    float* outPreds  = outLenDec + BB;                        // 128*29*10000
    float* outAlphas = outPreds + (size_t)BB * TM * VV;       // 128*29*196
    float* outOrder  = outAlphas + (size_t)BB * TM * PP;      // 128

    auto cvt = [&](const float* s, bf16_t* d, size_t n) {
        hipLaunchKernelGGL(k_cvt, dim3((unsigned)((n + 255) / 256)), dim3(256), 0, stream, s, d, n);
    };

    // weight conversions (bf16 weights stay resident in the 192MB L2 across the scan)
    cvt(W_enc_att, wEncAtt, (size_t)DECD * ENCD);
    cvt(W_dec_att, wDecAtt, (size_t)DECD * DECD);
    cvt(W_init_h,  wInitH,  (size_t)DECD * ENCD);
    cvt(W_init_c,  wInitC,  (size_t)DECD * ENCD);
    cvt(W_ih,      wIh,     (size_t)(4 * DECD) * (EMBD + ENCD));
    cvt(W_hh,      wHh,     (size_t)(4 * DECD) * DECD);
    cvt(W_fbeta,   wFbeta,  (size_t)ENCD * DECD);
    cvt(W_fc,      wFc,     (size_t)VV * DECD);

    hipLaunchKernelGGL(k_order, dim3(1), dim3(BB), 0, stream,
                       len_caps, enc_caps, ordW, lendecW, capsS, outCaps, outLenDec, outOrder);

    const size_t encTotal = (size_t)BB * PP * ENCD;
    hipLaunchKernelGGL(k_permcvt, dim3((unsigned)((encTotal + 255) / 256)), dim3(256), 0, stream,
                       enc_out, ordW, encBf, encTotal);
    hipLaunchKernelGGL(k_mu, dim3(ENCD / 256, BB), dim3(256), 0, stream, encBf, muBf);
    hipLaunchKernelGGL(k_embed, dim3((BB * TM * EMBD + 255) / 256), dim3(256), 0, stream,
                       emb, capsS, embeds);

    // h, c init: mu @ W_init_{h,c}^T + b
    launch_gemm(muBf, wInitH, b_init_h, hS, BB, DECD, ENCD, DECD, 0, 0, nullptr, 0, stream);
    launch_gemm(muBf, wInitC, b_init_c, cS, BB, DECD, ENCD, DECD, 0, 0, nullptr, 0, stream);
    cvt(hS, hbf, (size_t)BB * DECD);

    // att_enc = enc @ W_enc_att^T + b  (the big 52.7 GFLOP GEMM)
    launch_gemm(encBf, wEncAtt, b_enc_att, attEnc, BB * PP, DECD, ENCD, DECD, 0, 0, nullptr, 0, stream);

    for (int t = 0; t < TM; ++t) {
        // att_dec = h @ W_dec_att^T + b
        launch_gemm(hbf, wDecAtt, b_dec_att, attDec, BB, DECD, DECD, DECD, 0, 0, nullptr, 0, stream);
        // e = relu(att_enc + att_dec) . W_att + b_att
        hipLaunchKernelGGL(k_e, dim3(BB * PP / 8), dim3(256), 0, stream,
                           attEnc, attDec, W_att, b_att, eBuf);
        hipLaunchKernelGGL(k_softmax, dim3(BB), dim3(256), 0, stream,
                           eBuf, alpha, outAlphas, lendecW, t);
        hipLaunchKernelGGL(k_weighted, dim3(ENCD / 256, BB), dim3(256), 0, stream,
                           encBf, alpha, weighted);
        // gate = sigmoid(h @ W_fbeta^T + b_fbeta)
        launch_gemm(hbf, wFbeta, b_fbeta, gate, BB, ENCD, DECD, ENCD, 1, 0, nullptr, 0, stream);
        hipLaunchKernelGGL(k_wgx, dim3((EMBD + ENCD) / 256, BB), dim3(256), 0, stream,
                           embeds, weighted, gate, xbf, t);
        // gates = x @ W_ih^T + b_ih + h @ W_hh^T + b_hh
        launch_gemm(xbf, wIh, b_ih, gatesB, BB, 4 * DECD, EMBD + ENCD, 4 * DECD, 0, 0, nullptr, 0, stream);
        launch_gemm(hbf, wHh, b_hh, gatesB, BB, 4 * DECD, DECD, 4 * DECD, 0, 1, nullptr, 0, stream);
        hipLaunchKernelGGL(k_lstm, dim3(DECD / 256, BB), dim3(256), 0, stream,
                           gatesB, hS, cS, hbf, hnewbf, lendecW, t);
        // preds (masked) written directly into d_out with ldc = 29*V
        launch_gemm(hnewbf, wFc, b_fc, outPreds + (size_t)t * VV,
                    BB, VV, DECD, TM * VV, 0, 0, lendecW, t, stream);
    }
    (void)in_sizes; (void)n_in; (void)out_size; (void)ws_size;
}
